// PI05Attention_31181462569092
// MI455X (gfx1250) — matrix operand order repaired and checked by reading, not compiled, not run on hardware
//
#include <hip/hip_runtime.h>
#include <hip/hip_bf16.h>

typedef __attribute__((ext_vector_type(16))) _Float16 v16h;
typedef __attribute__((ext_vector_type(8)))  _Float16 v8h;
typedef __attribute__((ext_vector_type(8)))  float    v8f;

// ---------------- problem constants ----------------
#define BB     4
#define L_VLM  768
#define L_EXP  64
#define L_TOT  832
#define D_VLM  2048
#define D_EXP  1024
#define NH     8
#define DH     256
#define HD     2048          // NH*DH
#define SCALE  0.0625f       // 1/sqrt(256)

// ---------------- GEMM tile config ----------------
// All M,N in this problem are multiples of 64 and all K multiples of 32,
// so 64x64x32 tiles need no bounds checks anywhere.
#define TM 64
#define TN 64
#define TK 32
#define LDS_S (TK + 8)       // 40 f16 (80B row; keeps 16B alignment, skews banks)

// Async DMA: global -> LDS, 16 bytes per lane, tracked by ASYNCcnt.
__device__ __forceinline__ void async_copy_b128(void* lds_dst, const void* gsrc) {
  unsigned int l = (unsigned int)(unsigned long long)lds_dst;   // flat->LDS: addr[31:0]
  asm volatile("global_load_async_to_lds_b128 %0, %1, off"
               :: "v"(l), "v"(gsrc) : "memory");
}

// f32 -> f16 elementwise conversion
__global__ void cvt_f32_f16_kernel(const float* __restrict__ src,
                                   _Float16* __restrict__ dst, long n) {
  long i = (long)blockIdx.x * blockDim.x + threadIdx.x;
  if (i < n) dst[i] = (_Float16)src[i];
}

// f32 [K,N] -> f16 [N,K] transpose-convert (coalesced on the read side)
__global__ void cvtT_f32_f16_kernel(const float* __restrict__ src,
                                    _Float16* __restrict__ dst, int K, int N) {
  long idx = (long)blockIdx.x * blockDim.x + threadIdx.x;
  long total = (long)K * N;
  if (idx >= total) return;
  int n = (int)(idx % N);
  int k = (int)(idx / N);
  dst[(long)n * K + k] = (_Float16)src[idx];
}

// Batched WMMA GEMM:  C[z] = alpha * A[z] x B[z]^T_layout (+ addend)
//   A:  [M,K] row-major f16
//   Bt: [N,K] row-major f16 (i.e. logical B[K,N] stored transposed)
// Double-buffered LDS, async global->LDS staging (ASYNCcnt ping-pong),
// last k-iteration peeled so the hot loop is branch-free.
// Requires: M%64==0, N%64==0, K%32==0, lda%8==0, ldb%8==0.
__global__ __launch_bounds__(128)
void wmma_gemm_kernel(const _Float16* __restrict__ A, long strideA,
                      const _Float16* __restrict__ Bt, long strideB,
                      int M, int N, int K, int lda, int ldb,
                      float alpha,
                      const float* __restrict__ addend, int addDiv, long addStride, int ldadd,
                      float* __restrict__ Cf, _Float16* __restrict__ Ch,
                      int divC, long strideCo, long strideCi, int ldc)
{
  __shared__ _Float16 sA[2][TM * LDS_S];
  __shared__ _Float16 sB[2][TN * LDS_S];

  const int tid = threadIdx.x;      // 0..127 (4 waves)
  const int z   = blockIdx.z;
  const int m0  = blockIdx.y * TM;
  const int n0  = blockIdx.x * TN;

  const _Float16* Ab = A  + (long)z * strideA;
  const _Float16* Bb = Bt + (long)z * strideB;
  const long coff = (long)(z / divC) * strideCo + (long)(z % divC) * strideCi;
  float*    Cfb = Cf ? (Cf + coff) : nullptr;
  _Float16* Chb = Ch ? (Ch + coff) : nullptr;
  const float* Addb = addend ? (addend + (long)(z / addDiv) * addStride) : nullptr;

  const int lane = tid & 31;
  const int wave = tid >> 5;        // 0..3
  const int wm = (wave & 1) * 32;   // wave row offset inside tile
  const int wn = (wave >> 1) * 32;  // wave col offset inside tile

  // staging map: 64 rows x 4 chunks of 8 f16 = 256 chunks; 128 threads x 2
  const int r0 = tid >> 2;               // 0..31
  const int c0 = (tid & 3) * 8;          // 0/8/16/24

  v8f acc[2][2];
  #pragma unroll
  for (int i = 0; i < 2; ++i)
    #pragma unroll
    for (int j = 0; j < 2; ++j)
      #pragma unroll
      for (int e = 0; e < 8; ++e) acc[i][j][e] = 0.f;

  // issue async stage of buffer `buf` for k-offset k0 (4 async ops per wave)
  auto stage = [&](int buf, int k0) {
    #pragma unroll
    for (int c = 0; c < 2; ++c) {
      int row = r0 + c * 32;
      async_copy_b128(sA[buf] + row * LDS_S + c0,
                      Ab + (long)(m0 + row) * lda + k0 + c0);
      async_copy_b128(sB[buf] + row * LDS_S + c0,
                      Bb + (long)(n0 + row) * ldb + k0 + c0);
    }
  };

  // one k-step of compute from buffer `buf`
  auto compute = [&](int buf) {
    const int fm    = lane & 15;
    const int khalf = (lane >> 4) * 8;   // A: lane-half -> K 0-7 / 8-15 (and +16)
    const int bkh   = (lane >> 4) * 16;  // unused: B uses khalf (same K order as A)
    v16h afrag[2], bfrag[2];
    #pragma unroll
    for (int i = 0; i < 2; ++i) {
      const _Float16* ap = sA[buf] + (wm + i * 16 + fm) * LDS_S;
      union { v16h v; v8h h[2]; } u;
      u.h[0] = *(const v8h*)(ap + khalf);
      u.h[1] = *(const v8h*)(ap + 16 + khalf);
      afrag[i] = u.v;
    }
    #pragma unroll
    for (int i = 0; i < 2; ++i) {
      const _Float16* bp = sB[buf] + (wn + i * 16 + fm) * LDS_S;
      union { v16h v; v8h h[2]; } u;
      u.h[0] = *(const v8h*)(bp + khalf);
      u.h[1] = *(const v8h*)(bp + 16 + khalf);
      bfrag[i] = u.v;
    }
    #pragma unroll
    for (int i = 0; i < 2; ++i)
      #pragma unroll
      for (int j = 0; j < 2; ++j)
        acc[i][j] = __builtin_amdgcn_wmma_f32_16x16x32_f16(
            false, afrag[i], false, bfrag[j], (short)0, acc[i][j], false, false);
  };

  const int nIter = K / TK;
  stage(0, 0);
  int cur = 0;

  // main loop: unconditional stage(next) + wait(<=4); branch-free body
  for (int it = 0; it < nIter - 1; ++it) {
    stage(cur ^ 1, (it + 1) * TK);
    // async loads complete in order: <=4 outstanding => current buffer landed
    asm volatile("s_wait_asynccnt 0x4" ::: "memory");
    __syncthreads();
    compute(cur);
    __syncthreads();
    cur ^= 1;
  }
  // peeled last iteration
  asm volatile("s_wait_asynccnt 0x0" ::: "memory");
  __syncthreads();
  compute(cur);

  // ---- epilogue: C layout lanes 0-15 -> M=r, lanes 16-31 -> M=8+r ----
  const int cn  = lane & 15;
  const int cm8 = (lane >> 4) * 8;
  #pragma unroll
  for (int i = 0; i < 2; ++i)
    #pragma unroll
    for (int j = 0; j < 2; ++j)
      #pragma unroll
      for (int r = 0; r < 8; ++r) {
        int m = m0 + wm + i * 16 + cm8 + r;
        int n = n0 + wn + j * 16 + cn;
        float v = acc[i][j][r] * alpha;
        if (Addb) v += Addb[(long)m * ldadd + n];
        if (Cfb) Cfb[(long)m * ldc + n] = v;
        if (Chb) Chb[(long)m * ldc + n] = (_Float16)v;
      }
}

// Pack projections into q,k [B,H,L,DH] (RoPE applied) and v^T [B,H,DH,L].
__global__ void pack_rope_kernel(const _Float16* __restrict__ Qv, const _Float16* __restrict__ Qe,
                                 const _Float16* __restrict__ Kv, const _Float16* __restrict__ Ke,
                                 const _Float16* __restrict__ Vv, const _Float16* __restrict__ Ve,
                                 const int* __restrict__ pos_ids,
                                 _Float16* __restrict__ q, _Float16* __restrict__ k,
                                 _Float16* __restrict__ vT)
{
  long idx = (long)blockIdx.x * blockDim.x + threadIdx.x;   // over B*L_TOT*H*128
  const long total = (long)BB * L_TOT * NH * (DH / 2);
  if (idx >= total) return;
  int d2 = (int)(idx & 127);
  int h  = (int)((idx >> 7) & (NH - 1));
  long rest = idx >> 10;
  int l = (int)(rest % L_TOT);
  int b = (int)(rest / L_TOT);

  const _Float16 *Qs, *Ks, *Vs; long srcoff;
  if (l < L_VLM) {
    srcoff = ((long)(b * L_VLM + l)) * HD + h * DH;
    Qs = Qv; Ks = Kv; Vs = Vv;
  } else {
    srcoff = ((long)(b * L_EXP + (l - L_VLM))) * HD + h * DH;
    Qs = Qe; Ks = Ke; Vs = Ve;
  }
  float pos = (float)pos_ids[b * L_TOT + l];
  float inv_freq = __expf(-__logf(10000.0f) * (float)(2 * d2) / (float)DH);
  float fr = pos * inv_freq;
  float c = __cosf(fr), sn = __sinf(fr);

  long bh  = (long)(b * NH + h);
  long dst = (bh * L_TOT + l) * DH + d2;
  float q1 = (float)Qs[srcoff + d2], q2 = (float)Qs[srcoff + 128 + d2];
  q[dst]       = (_Float16)(q1 * c - q2 * sn);
  q[dst + 128] = (_Float16)(q2 * c + q1 * sn);
  float k1 = (float)Ks[srcoff + d2], k2 = (float)Ks[srcoff + 128 + d2];
  k[dst]       = (_Float16)(k1 * c - k2 * sn);
  k[dst + 128] = (_Float16)(k2 * c + k1 * sn);
  // v stored transposed: [bh, d, l]
  vT[(bh * DH + d2)       * L_TOT + l] = Vs[srcoff + d2];
  vT[(bh * DH + d2 + 128) * L_TOT + l] = Vs[srcoff + 128 + d2];
}

// Row softmax: f32 scores -> f16 probs. One 256-thread block per row of length L.
__global__ __launch_bounds__(256)
void softmax_kernel(const float* __restrict__ S, _Float16* __restrict__ P, int L)
{
  __shared__ float red[256];
  long row = blockIdx.x;
  const float* s = S + row * (long)L;
  int tid = threadIdx.x;

  float lmax = -3.4e38f;
  for (int i = tid; i < L; i += 256) lmax = fmaxf(lmax, s[i]);
  red[tid] = lmax; __syncthreads();
  for (int st = 128; st > 0; st >>= 1) {
    if (tid < st) red[tid] = fmaxf(red[tid], red[tid + st]);
    __syncthreads();
  }
  float m = red[0]; __syncthreads();

  float lsum = 0.f;
  for (int i = tid; i < L; i += 256) lsum += __expf(s[i] - m);
  red[tid] = lsum; __syncthreads();
  for (int st = 128; st > 0; st >>= 1) {
    if (tid < st) red[tid] += red[tid + st];
    __syncthreads();
  }
  float inv = 1.f / red[0];
  for (int i = tid; i < L; i += 256)
    P[row * (long)L + i] = (_Float16)(__expf(s[i] - m) * inv);
}

extern "C" void kernel_launch(void* const* d_in, const int* in_sizes, int n_in,
                              void* d_out, int out_size, void* d_ws, size_t ws_size,
                              hipStream_t stream)
{
  (void)in_sizes; (void)n_in; (void)out_size; (void)ws_size;
  const float* vlm_hidden = (const float*)d_in[0];
  const float* exp_hidden = (const float*)d_in[1];
  const float* attn_mask  = (const float*)d_in[2];
  const int*   pos_ids    = (const int*)d_in[3];
  const float* wq_v = (const float*)d_in[4];
  const float* wk_v = (const float*)d_in[5];
  const float* wv_v = (const float*)d_in[6];
  const float* wo_v = (const float*)d_in[7];
  const float* wq_e = (const float*)d_in[8];
  const float* wk_e = (const float*)d_in[9];
  const float* wv_e = (const float*)d_in[10];
  const float* wo_e = (const float*)d_in[11];

  // ---- carve workspace ----
  char* p = (char*)d_ws;
  auto takeH = [&](long n) { _Float16* r = (_Float16*)p; p += ((n * 2 + 255) / 256) * 256; return r; };
  auto takeF = [&](long n) { float* r = (float*)p;       p += ((n * 4 + 255) / 256) * 256; return r; };

  _Float16* hv   = takeH((long)BB * L_VLM * D_VLM);
  _Float16* he   = takeH((long)BB * L_EXP * D_EXP);
  _Float16* fwqv = takeH((long)D_VLM * HD);   // all weights stored [N,K]
  _Float16* fwkv = takeH((long)D_VLM * HD);
  _Float16* fwvv = takeH((long)D_VLM * HD);
  _Float16* fwov = takeH((long)HD * D_VLM);
  _Float16* fwqe = takeH((long)D_EXP * HD);
  _Float16* fwke = takeH((long)D_EXP * HD);
  _Float16* fwve = takeH((long)D_EXP * HD);
  _Float16* fwoe = takeH((long)HD * D_EXP);
  _Float16* Qv   = takeH((long)BB * L_VLM * HD);
  _Float16* Kv   = takeH((long)BB * L_VLM * HD);
  _Float16* Vv   = takeH((long)BB * L_VLM * HD);
  _Float16* Qe   = takeH((long)BB * L_EXP * HD);
  _Float16* Ke   = takeH((long)BB * L_EXP * HD);
  _Float16* Ve   = takeH((long)BB * L_EXP * HD);
  _Float16* qpk  = takeH((long)BB * NH * L_TOT * DH);
  _Float16* kpk  = takeH((long)BB * NH * L_TOT * DH);
  _Float16* vTpk = takeH((long)BB * NH * DH * L_TOT);
  _Float16* ctx  = takeH((long)BB * L_TOT * HD);
  _Float16* prob = takeH((long)BB * NH * L_TOT * L_TOT);
  float*    scor = takeF((long)BB * NH * L_TOT * L_TOT);

  auto cvt = [&](const float* s, _Float16* d, long n) {
    cvt_f32_f16_kernel<<<dim3((unsigned)((n + 255) / 256)), dim3(256), 0, stream>>>(s, d, n);
  };
  auto cvtT = [&](const float* s, _Float16* d, int K, int N) {
    long n = (long)K * N;
    cvtT_f32_f16_kernel<<<dim3((unsigned)((n + 255) / 256)), dim3(256), 0, stream>>>(s, d, K, N);
  };
  auto gemm = [&](const _Float16* A, long sA, const _Float16* Bt, long sB,
                  int M, int N, int K, int lda, int ldb, float alpha,
                  const float* add, int addDiv, long addStride, int ldadd,
                  float* Cf, _Float16* Ch, int divC, long sCo, long sCi, int ldc, int batch) {
    dim3 grid((unsigned)(N / TN), (unsigned)(M / TM), (unsigned)batch);
    wmma_gemm_kernel<<<grid, dim3(128), 0, stream>>>(A, sA, Bt, sB, M, N, K, lda, ldb,
                                                     alpha, add, addDiv, addStride, ldadd,
                                                     Cf, Ch, divC, sCo, sCi, ldc);
  };

  // 1) convert hiddens; transpose-convert all weights to [N,K] f16
  cvt(vlm_hidden, hv, (long)BB * L_VLM * D_VLM);
  cvt(exp_hidden, he, (long)BB * L_EXP * D_EXP);
  cvtT(wq_v, fwqv, D_VLM, HD);  cvtT(wk_v, fwkv, D_VLM, HD);
  cvtT(wv_v, fwvv, D_VLM, HD);  cvtT(wo_v, fwov, HD, D_VLM);
  cvtT(wq_e, fwqe, D_EXP, HD);  cvtT(wk_e, fwke, D_EXP, HD);
  cvtT(wv_e, fwve, D_EXP, HD);  cvtT(wo_e, fwoe, HD, D_EXP);

  // 2) QKV projections (plain [B*L, 2048] f16 outputs)
  gemm(hv, 0, fwqv, 0, BB * L_VLM, HD, D_VLM, D_VLM, D_VLM, 1.f, nullptr, 1, 0, 1,
       nullptr, Qv, 1, 0, 0, HD, 1);
  gemm(hv, 0, fwkv, 0, BB * L_VLM, HD, D_VLM, D_VLM, D_VLM, 1.f, nullptr, 1, 0, 1,
       nullptr, Kv, 1, 0, 0, HD, 1);
  gemm(hv, 0, fwvv, 0, BB * L_VLM, HD, D_VLM, D_VLM, D_VLM, 1.f, nullptr, 1, 0, 1,
       nullptr, Vv, 1, 0, 0, HD, 1);
  gemm(he, 0, fwqe, 0, BB * L_EXP, HD, D_EXP, D_EXP, D_EXP, 1.f, nullptr, 1, 0, 1,
       nullptr, Qe, 1, 0, 0, HD, 1);
  gemm(he, 0, fwke, 0, BB * L_EXP, HD, D_EXP, D_EXP, D_EXP, 1.f, nullptr, 1, 0, 1,
       nullptr, Ke, 1, 0, 0, HD, 1);
  gemm(he, 0, fwve, 0, BB * L_EXP, HD, D_EXP, D_EXP, D_EXP, 1.f, nullptr, 1, 0, 1,
       nullptr, Ve, 1, 0, 0, HD, 1);

  // 3) concat + head split + RoPE -> q,k [B,H,832,256]; v^T [B,H,256,832]
  {
    long total = (long)BB * L_TOT * NH * (DH / 2);
    pack_rope_kernel<<<dim3((unsigned)((total + 255) / 256)), dim3(256), 0, stream>>>(
        Qv, Qe, Kv, Ke, Vv, Ve, pos_ids, qpk, kpk, vTpk);
  }

  // 4) scores = Q K^T * SCALE + mask  (kpk already [N=keys, K=DH] layout)
  gemm(qpk, (long)L_TOT * DH, kpk, (long)L_TOT * DH,
       L_TOT, L_TOT, DH, DH, DH, SCALE,
       attn_mask, NH, (long)L_TOT * L_TOT, L_TOT,
       scor, nullptr, 1, (long)L_TOT * L_TOT, 0, L_TOT, BB * NH);

  // 5) softmax rows -> f16 probs
  softmax_kernel<<<dim3(BB * NH * L_TOT), dim3(256), 0, stream>>>(scor, prob, L_TOT);

  // 6) ctx = P V  (B side = vT [DH, L]), scattered to [B, L, H*DH]
  gemm(prob, (long)L_TOT * L_TOT, vTpk, (long)DH * L_TOT,
       L_TOT, DH, L_TOT, L_TOT, L_TOT, 1.f, nullptr, 1, 0, 1,
       nullptr, ctx, NH, (long)L_TOT * HD, DH, HD, BB * NH);

  // 7) output projections -> d_out (vlm flat, then exp flat)
  gemm(ctx, (long)L_TOT * HD, fwov, 0, L_VLM, D_VLM, HD, HD, HD, 1.f,
       nullptr, 1, 0, 1, (float*)d_out, nullptr, 1, (long)L_VLM * D_VLM, 0, D_VLM, BB);
  gemm(ctx + (long)L_VLM * HD, (long)L_TOT * HD, fwoe, 0, L_EXP, D_EXP, HD, HD, HD, 1.f,
       nullptr, 1, 0, 1, (float*)d_out + (long)BB * L_VLM * D_VLM, nullptr,
       1, (long)L_EXP * D_EXP, 0, D_EXP, BB);
}
